// MSCN_31671088840803
// MI455X (gfx1250) — matrix operand order repaired and checked by reading, not compiled, not run on hardware
//
#include <hip/hip_runtime.h>
#include <hip/hip_bf16.h>

// ---------------------------------------------------------------------------
// MI455X (gfx1250) multi-scale pooled cross-attention.
// GEMMs: v_wmma_f32_16x16x32_bf16 (wave32 WMMA).
// B-operand tiles staged to LDS via the Tensor Data Mover (tensor_load_to_lds,
// TENSORcnt-waited), read back as ds_load fragments.
// ---------------------------------------------------------------------------

typedef __bf16 bf16_t;
typedef __attribute__((ext_vector_type(16))) __bf16 v16bf;
typedef __attribute__((ext_vector_type(8)))  float  v8f;
typedef __attribute__((ext_vector_type(4)))  unsigned int u32x4;
typedef __attribute__((ext_vector_type(4)))  int i32x4;
typedef __attribute__((ext_vector_type(8)))  int i32x8;

#define B_     64
#define C_     512
#define N_     49      // 7*7
#define N1_    784     // 28*28
#define HEADS_ 8
#define HD_    64
#define SCALE_ 0.125f  // 64^-0.5
#define KC1_   392     // 784 // 2
#define KC2_   261     // 784 // 3
#define LN_EPS_ 1e-5f

#define RQ_    3136    // B_*N_
#define RQPAD_ 3200    // padded to multiple of 128
#define RKV_   50176   // B_*N1_

#if __has_builtin(__builtin_amdgcn_tensor_load_to_lds)
#define MSCN_HAS_TDM 1
#else
#define MSCN_HAS_TDM 0
#endif

extern __shared__ bf16_t lds_dyn[];      // dynamic LDS: staged B tiles

// ---------------- WMMA helpers (CDNA5 16x16x32 bf16 layouts) ----------------

__device__ inline v8f wmma_bf16(v16bf a, v16bf b, v8f c) {
  // 8 args: (neg_a, A, neg_b, B, c_mod, C, reuse_a, reuse_b)
  return __builtin_amdgcn_wmma_f32_16x16x32_bf16(false, a, false, b, (short)0, c,
                                                 false, false);
}

__device__ inline v8f zero8() {
  v8f z = {0.f, 0.f, 0.f, 0.f, 0.f, 0.f, 0.f, 0.f};
  return z;
}

// A (16x32 bf16, row-major source [.., stride]):
// lane 0-15 : M = m0+lane, elems 0-7 -> K = k0+0..7,  elems 8-15 -> K = k0+16..23
// lane 16-31: M = m0+lane-16, elems 0-7 -> K = k0+8..15, elems 8-15 -> K = k0+24..31
__device__ inline v16bf load_a_frag(const bf16_t* base, int stride, int m0, int k0,
                                    int lane) {
  const bf16_t* p = base + (size_t)(m0 + (lane & 15)) * stride + k0
                  + ((lane >> 4) & 1) * 8;
  v16bf a;
#pragma unroll
  for (int i = 0; i < 8; ++i) { a[i] = p[i]; a[i + 8] = p[i + 16]; }
  return a;
}

// B (32x16 bf16) from a column-major ("transposed") buffer BT[col][k]:
// lane 0-15 : N = n0+lane, elems 0-7 -> K = k0+0..7,  elems 8-15 -> K = k0+16..23
// lane 16-31: N = n0+lane-16, elems 0-7 -> K = k0+8..15, elems 8-15 -> K = k0+24..31
__device__ inline v16bf load_b_frag(const bf16_t* baseT, int stride, int n0, int k0,
                                    int lane) {
  const bf16_t* p = baseT + (size_t)(n0 + (lane & 15)) * stride + k0
                  + ((lane >> 4) & 1) * 8;
  v16bf b;
#pragma unroll
  for (int e = 0; e < 16; ++e) b[e] = p[e + (e & 8)];
  return b;
}

// Same with K-range guard (K = 784 -> last 32-tile half valid).
__device__ inline v16bf load_b_frag_g(const bf16_t* baseT, int stride, int n0, int k0,
                                      int lane, int kmax) {
  int kb = k0 + ((lane >> 4) & 1) * 8;
  const bf16_t* p = baseT + (size_t)(n0 + (lane & 15)) * stride + kb;
  v16bf b;
#pragma unroll
  for (int e = 0; e < 16; ++e) b[e] = (kb + e + (e & 8) < kmax) ? p[e + (e & 8)] : (bf16_t)0.f;
  return b;
}

// ---------------- TDM staging: contiguous global region -> LDS -------------

__device__ inline unsigned dyn_lds_offset() {
#if __has_builtin(__builtin_amdgcn_groupstaticsize)
  return (unsigned)__builtin_amdgcn_groupstaticsize();
#else
  return 0u;
#endif
}

#if MSCN_HAS_TDM
// 1-D tensor DMA descriptor (ISA 8.3/8.4): data_size=2B, one row of n2 elems.
__device__ inline void tdm_load_1d(unsigned lds_off, const void* gsrc,
                                   unsigned n2 /* #bf16 elems, <= 65535 */) {
  unsigned long long ga = (unsigned long long)(uintptr_t)gsrc;
  u32x4 g0;
  g0[0] = 1u;                                            // count=1 user desc
  g0[1] = lds_off;                                       // lds_addr (bytes)
  g0[2] = (unsigned)(ga & 0xffffffffull);                // global_addr[31:0]
  g0[3] = (unsigned)((ga >> 32) & 0x01ffffffull)         // global_addr[56:32]
        | (2u << 30);                                    // type = 2 ("image")
  i32x8 g1;
  g1[0] = (int)(1u << 16);                               // data_size=1 (2B)
  g1[1] = (int)((n2 & 0xffffu) << 16);                   // tensor_dim0[15:0]
  g1[2] = (int)(((n2 >> 16) & 0xffffu) | (1u << 16));    // td0 hi | tensor_dim1=1
  g1[3] = (int)((n2 & 0xffffu) << 16);                   // tile_dim0 = n2
  g1[4] = 1;                                             // tile_dim1=1, tile_dim2=0
  g1[5] = (int)n2;                                       // tensor_dim0_stride lo
  g1[6] = 0;
  g1[7] = 0;
  i32x4 gz = {0, 0, 0, 0};
#if defined(__clang_major__) && (__clang_major__ >= 23)
  i32x8 gz8 = {0, 0, 0, 0, 0, 0, 0, 0};
  __builtin_amdgcn_tensor_load_to_lds(g0, g1, gz, gz, gz8, 0);
#else
  __builtin_amdgcn_tensor_load_to_lds(g0, g1, gz, gz, 0);
#endif
}
#endif

// Stage `bytes` (16B multiple) from contiguous global memory into LDS.
__device__ inline void stage_tile(bf16_t* lds, const bf16_t* g, unsigned bytes) {
#if MSCN_HAS_TDM
  if (threadIdx.x < 32) {                 // one TDM op from wave 0 (EXEC ignored)
    tdm_load_1d(dyn_lds_offset(), g, bytes >> 1);
    __builtin_amdgcn_s_wait_tensorcnt(0);
  }
  __syncthreads();
#else
  for (unsigned off = threadIdx.x * 16u; off < bytes; off += 256u * 16u) {
    *(uint4*)((char*)lds + off) = *(const uint4*)((const char*)g + off);
  }
  __syncthreads();
#endif
}

// ---------------- block reductions -----------------------------------------

__device__ inline float block_sum(float v, float* red) {
  int t = threadIdx.x;
  red[t] = v; __syncthreads();
#pragma unroll
  for (int o = 128; o > 0; o >>= 1) {
    if (t < o) red[t] += red[t + o];
    __syncthreads();
  }
  float r = red[0]; __syncthreads();
  return r;
}

__device__ inline float block_max(float v, float* red) {
  int t = threadIdx.x;
  red[t] = v; __syncthreads();
#pragma unroll
  for (int o = 128; o > 0; o >>= 1) {
    if (t < o) red[t] = fmaxf(red[t], red[t + o]);
    __syncthreads();
  }
  float r = red[0]; __syncthreads();
  return r;
}

__device__ inline float block_min(float v, float* red) {
  int t = threadIdx.x;
  red[t] = v; __syncthreads();
#pragma unroll
  for (int o = 128; o > 0; o >>= 1) {
    if (t < o) red[t] = fminf(red[t], red[t + o]);
    __syncthreads();
  }
  float r = red[0]; __syncthreads();
  return r;
}

// ---------------- 1. prep: weight transpose -> bf16, x transpose -----------

__global__ __launch_bounds__(256) void mscn_prep_kernel(
    const float* __restrict__ Wq, const float* __restrict__ Wkv,
    const float* __restrict__ Wproj, const float* __restrict__ x,
    bf16_t* __restrict__ WqT, bf16_t* __restrict__ WkvT,
    bf16_t* __restrict__ WprojT, bf16_t* __restrict__ xs) {
  int idx = blockIdx.x * 256 + threadIdx.x;
  const int NW1 = 512 * 512, NW2 = 512 * 1024, NW3 = 512 * 512, NX = RQ_ * 512;
  if (idx < NW1) {                        // Wq [in][out] -> WqT [out][in]
    int o = idx & 511, i = idx >> 9;
    WqT[(size_t)o * 512 + i] = (bf16_t)Wq[idx];
    return;
  }
  idx -= NW1;
  if (idx < NW2) {                        // Wkv [in][2C] -> WkvT [2C][in]
    int o = idx & 1023, i = idx >> 10;
    WkvT[(size_t)o * 512 + i] = (bf16_t)Wkv[idx];
    return;
  }
  idx -= NW2;
  if (idx < NW3) {                        // Wproj [in][out] -> WprojT [out][in]
    int o = idx & 511, i = idx >> 9;
    WprojT[(size_t)o * 512 + i] = (bf16_t)Wproj[idx];
    return;
  }
  idx -= NW3;
  if (idx < NX) {                         // x [b][c][n] -> xs [b*49+n][c]
    int n = idx % 49;
    int bc = idx / 49;
    int b = bc >> 9, c = bc & 511;
    xs[((size_t)b * 49 + n) * 512 + c] = (bf16_t)x[idx];
  }
}

// ---------------- 2. multi-scale avg pool (3/5/7, same, count_include_pad) --

__global__ __launch_bounds__(256) void mscn_pool_kernel(
    const float* __restrict__ y, float* __restrict__ pooled) {
  __shared__ float P[29][30];             // exclusive 2D integral image
  int bc = blockIdx.x;                    // b*512 + c
  int b = bc >> 9, c = bc & 511;
  const float* plane = y + (size_t)bc * N1_;
  int t = threadIdx.x;

  for (int i = t; i < 29 * 29; i += 256) {
    int r = i / 29, cc = i % 29;
    P[r][cc] = (r > 0 && cc > 0) ? plane[(r - 1) * 28 + (cc - 1)] : 0.f;
  }
  __syncthreads();
  if (t < 29) {                           // prefix along rows
    float s = 0.f;
    for (int j = 0; j < 29; ++j) { s += P[t][j]; P[t][j] = s; }
  }
  __syncthreads();
  if (t < 29) {                           // prefix along columns
    float s = 0.f;
    for (int i = 0; i < 29; ++i) { s += P[i][t]; P[i][t] = s; }
  }
  __syncthreads();

  for (int i = t; i < N1_; i += 256) {
    int r = i / 28, cc = i % 28;
    float acc = 0.f;
#pragma unroll
    for (int kk = 0; kk < 3; ++kk) {
      int p = kk + 1;                     // pad 1,2,3 ; kernel 3,5,7
      int k = 2 * p + 1;
      int r1 = max(r - p, 0), r2 = min(r + p + 1, 28);
      int c1 = max(cc - p, 0), c2 = min(cc + p + 1, 28);
      float s = P[r2][c2] - P[r1][c2] - P[r2][c1] + P[r1][c1];
      acc += s / (float)(k * k);
    }
    // write transposed so LayerNorm reads contiguous C
    pooled[((size_t)b * N1_ + i) * C_ + c] = acc;
  }
}

// ---------------- 3. LayerNorm over C -> bf16 -------------------------------

__global__ __launch_bounds__(256) void mscn_ln_kernel(
    const float* __restrict__ pooled, const float* __restrict__ gamma,
    const float* __restrict__ beta, bf16_t* __restrict__ yn) {
  __shared__ float red[256];
  size_t row = blockIdx.x;                // b*784 + n1
  const float* src = pooled + row * C_;
  int t = threadIdx.x;
  float v0 = src[t], v1 = src[t + 256];
  float mu = block_sum(v0 + v1, red) * (1.f / 512.f);
  float d0 = v0 - mu, d1 = v1 - mu;
  float var = block_sum(d0 * d0 + d1 * d1, red) * (1.f / 512.f);
  float rs = rsqrtf(var + LN_EPS_);
  bf16_t* dst = yn + row * C_;
  dst[t]       = (bf16_t)(d0 * rs * gamma[t] + beta[t]);
  dst[t + 256] = (bf16_t)(d1 * rs * gamma[t + 256] + beta[t + 256]);
}

// ---------------- 4. kv GEMM: yn[50176,512] @ WkvT -> K / V^T (bf16) --------
// B tile (64 cols x 512 k = 64KB) TDM-staged to LDS once per block.

__global__ __launch_bounds__(256) void mscn_kv_gemm_kernel(
    const bf16_t* __restrict__ yn, const bf16_t* __restrict__ WkvT,
    bf16_t* __restrict__ Kbuf, bf16_t* __restrict__ vT) {
  int lane = threadIdx.x & 31, wave = threadIdx.x >> 5;
  int m0 = blockIdx.x * 128 + wave * 16;  // 392 row-blocks
  int n0 = blockIdx.y * 64;               // 16 col-blocks

  stage_tile(lds_dyn, WkvT + (size_t)n0 * 512, 64 * 512 * 2);

  v8f acc[4] = {zero8(), zero8(), zero8(), zero8()};
#pragma unroll 4
  for (int k0 = 0; k0 < 512; k0 += 32) {
    __builtin_prefetch(yn + (size_t)(m0 + (lane & 15)) * 512 + k0 + 64, 0, 0);
    v16bf a = load_a_frag(yn, 512, m0, k0, lane);
#pragma unroll
    for (int nt = 0; nt < 4; ++nt) {
      v16bf b = load_b_frag(lds_dyn, 512, nt * 16, k0, lane);
      acc[nt] = wmma_bf16(a, b, acc[nt]);
    }
  }
#pragma unroll
  for (int nt = 0; nt < 4; ++nt) {
#pragma unroll
    for (int r = 0; r < 8; ++r) {
      int row = m0 + r + ((lane >> 4) << 3);
      int col = n0 + nt * 16 + (lane & 15);
      int b = row / N1_, n1 = row % N1_;
      bf16_t val = (bf16_t)acc[nt][r];
      if (col < 512) {                    // K: [b][h][n1][d] (d contiguous)
        int head = col >> 6, dd = col & 63;
        Kbuf[(((size_t)b * 8 + head) * N1_ + n1) * 64 + dd] = val;
      } else {                            // V^T: [b][h][d][n1] (n1 contiguous)
        int cv = col - 512;
        int head = cv >> 6, dd = cv & 63;
        vT[(((size_t)b * 8 + head) * 64 + dd) * N1_ + n1] = val;
      }
    }
  }
}

// ---------------- 5. q GEMM: xs[3136,512] @ WqT, scale folded ---------------

__global__ __launch_bounds__(256) void mscn_q_gemm_kernel(
    const bf16_t* __restrict__ xs, const bf16_t* __restrict__ WqT,
    bf16_t* __restrict__ qws) {
  int lane = threadIdx.x & 31, wave = threadIdx.x >> 5;
  int m0 = blockIdx.x * 128 + wave * 16;  // 25 row-blocks (rows padded to 3200)
  int n0 = blockIdx.y * 64;               // 8 col-blocks

  stage_tile(lds_dyn, WqT + (size_t)n0 * 512, 64 * 512 * 2);

  v8f acc[4] = {zero8(), zero8(), zero8(), zero8()};
#pragma unroll 4
  for (int k0 = 0; k0 < 512; k0 += 32) {
    v16bf a = load_a_frag(xs, 512, m0, k0, lane);
#pragma unroll
    for (int nt = 0; nt < 4; ++nt) {
      v16bf b = load_b_frag(lds_dyn, 512, nt * 16, k0, lane);
      acc[nt] = wmma_bf16(a, b, acc[nt]);
    }
  }
#pragma unroll
  for (int nt = 0; nt < 4; ++nt) {
#pragma unroll
    for (int r = 0; r < 8; ++r) {
      int row = m0 + r + ((lane >> 4) << 3);
      if (row < RQ_) {
        int col = n0 + nt * 16 + (lane & 15);
        int b = row / 49, n = row % 49;
        int head = col >> 6, dd = col & 63;
        // padded layout [b][h][64][64]; pad rows pre-zeroed by memset
        qws[(((size_t)b * 8 + head) * 64 + n) * 64 + dd] =
            (bf16_t)(acc[nt][r] * SCALE_);
      }
    }
  }
}

// ---------------- 6. scores: S[bh][64][784] = q @ K^T -----------------------
// Whole K tile for (b,h) (784x64 bf16 = 98KB) TDM-staged to LDS.

__global__ __launch_bounds__(256) void mscn_score_kernel(
    const bf16_t* __restrict__ qws, const bf16_t* __restrict__ Kbuf,
    float* __restrict__ S) {
  int bh = blockIdx.x;                    // 512 blocks
  const bf16_t* q  = qws  + (size_t)bh * 64 * 64;
  const bf16_t* kb = Kbuf + (size_t)bh * N1_ * 64;
  float* Sb = S + (size_t)bh * 64 * N1_;
  int lane = threadIdx.x & 31, wave = threadIdx.x >> 5;

  stage_tile(lds_dyn, kb, N1_ * 64 * 2);

  v16bf aq[4][2];                         // whole q tile in registers
#pragma unroll
  for (int mt = 0; mt < 4; ++mt)
#pragma unroll
    for (int ks = 0; ks < 2; ++ks)
      aq[mt][ks] = load_a_frag(q, 64, mt * 16, ks * 32, lane);

  for (int nt = wave; nt < 49; nt += 8) { // 49 N1-tiles across 8 waves
    int n0 = nt * 16;
    v16bf b0 = load_b_frag(lds_dyn, 64, n0, 0,  lane);
    v16bf b1 = load_b_frag(lds_dyn, 64, n0, 32, lane);
#pragma unroll
    for (int mt = 0; mt < 4; ++mt) {
      v8f acc = zero8();
      acc = wmma_bf16(aq[mt][0], b0, acc);
      acc = wmma_bf16(aq[mt][1], b1, acc);
#pragma unroll
      for (int r = 0; r < 8; ++r) {
        int row = mt * 16 + r + ((lane >> 4) << 3);
        int col = n0 + (lane & 15);
        Sb[(size_t)row * N1_ + col] = acc[r];
      }
    }
  }
}

// ---------------- 7. per-row top-k thresholds + softmax stats ---------------
// stats[row*8] = {rowmax, thr_top392, thr_top261, w1/sum1, w2/sum2}

__global__ __launch_bounds__(256) void mscn_stats_kernel(
    const float* __restrict__ S, const float* __restrict__ w1p,
    const float* __restrict__ w2p, float* __restrict__ stats) {
  __shared__ float srow[N1_];
  __shared__ float red[256];
  int rowid = blockIdx.x;                 // 0 .. B*h*49-1
  int bh = rowid / 49, m = rowid % 49;
  const float* rowp = S + ((size_t)bh * 64 + m) * N1_;
  int t = threadIdx.x;

  for (int i = t; i < N1_; i += 256) srow[i] = rowp[i];
  __syncthreads();

  float lmax = -3.4e38f, lmin = 3.4e38f;
  for (int i = t; i < N1_; i += 256) {
    float v = srow[i];
    lmax = fmaxf(lmax, v);
    lmin = fminf(lmin, v);
  }
  float mx = block_max(lmax, red);
  float mn = block_min(lmin, red);

  // bisection select for kth-largest (continuous logits => converges)
  const int kc[2] = {KC1_, KC2_};
  float thr[2];
#pragma unroll
  for (int j = 0; j < 2; ++j) {
    float lo = mn, hi = mx;
    for (int it = 0; it < 28; ++it) {
      float mid = 0.5f * (lo + hi);
      float cl = 0.f;
      for (int i = t; i < N1_; i += 256) cl += (srow[i] >= mid) ? 1.f : 0.f;
      float cnt = block_sum(cl, red);
      if (cnt >= (float)kc[j]) lo = mid; else hi = mid;
    }
    thr[j] = lo;
  }

  float s1 = 0.f, s2 = 0.f;
  for (int i = t; i < N1_; i += 256) {
    float v = srow[i];
    float e = __expf(v - mx);
    if (v >= thr[0]) s1 += e;
    if (v >= thr[1]) s2 += e;
  }
  float sum1 = block_sum(s1, red);
  float sum2 = block_sum(s2, red);
  if (t == 0) {
    float* o = stats + (size_t)rowid * 8;
    o[0] = mx;
    o[1] = thr[0];
    o[2] = thr[1];
    o[3] = w1p[0] / sum1;
    o[4] = w2p[0] / sum2;
  }
}

// ---------------- 8. fused (w1*P1 + w2*P2) @ V ------------------------------
// V^T tile (64x784 bf16 = 98KB) TDM-staged to LDS; P built on the fly.

__device__ inline bf16_t p_elem(const float* srow, int k, float mx, float t1,
                                float t2, float c1, float c2) {
  if (k >= N1_) return (bf16_t)0.f;
  float s = srow[k];
  float e = __expf(s - mx);
  float p = 0.f;
  if (s >= t1) p += c1 * e;
  if (s >= t2) p += c2 * e;
  return (bf16_t)p;
}

__global__ __launch_bounds__(256) void mscn_av_kernel(
    const float* __restrict__ S, const float* __restrict__ stats,
    const bf16_t* __restrict__ vT, bf16_t* __restrict__ outw) {
  int bh = blockIdx.x;                    // 512 blocks
  int b = bh >> 3, h = bh & 7;
  const float*  Sb  = S  + (size_t)bh * 64 * N1_;
  const bf16_t* vtb = vT + (size_t)bh * 64 * N1_;
  int lane = threadIdx.x & 31, wave = threadIdx.x >> 5;
  int mt  = wave & 3;                     // M tile 0..3
  int nt0 = (wave >> 2) * 2;              // N tiles {0,1} or {2,3}
  int nt1 = nt0 + 1;

  stage_tile(lds_dyn, vtb, 64 * N1_ * 2);

  int arow = mt * 16 + (lane & 15);       // row this lane supplies for A
  float mx = 0.f, t1 = 0.f, t2 = 0.f, c1 = 0.f, c2 = 0.f;
  if (arow < 49) {                        // pad rows contribute zero
    const float* st = stats + ((size_t)bh * 49 + arow) * 8;
    mx = st[0]; t1 = st[1]; t2 = st[2]; c1 = st[3]; c2 = st[4];
  }
  const float* srow = Sb + (size_t)arow * N1_;

  v8f acc0 = zero8(), acc1 = zero8();
  for (int ks = 0; ks < 25; ++ks) {       // K = 784 -> 24.5 tiles of 32
    int kbase = ks * 32 + ((lane >> 4) << 3);
    v16bf a;
#pragma unroll
    for (int i = 0; i < 8; ++i) {
      a[i]     = p_elem(srow, kbase + i,      mx, t1, t2, c1, c2);
      a[i + 8] = p_elem(srow, kbase + 16 + i, mx, t1, t2, c1, c2);
    }
    v16bf b0 = load_b_frag_g(lds_dyn, N1_, nt0 * 16, ks * 32, lane, N1_);
    v16bf b1 = load_b_frag_g(lds_dyn, N1_, nt1 * 16, ks * 32, lane, N1_);
    acc0 = wmma_bf16(a, b0, acc0);
    acc1 = wmma_bf16(a, b1, acc1);
  }
#pragma unroll
  for (int r = 0; r < 8; ++r) {
    int row = mt * 16 + r + ((lane >> 4) << 3);
    if (row < 49) {
      int col0 = h * 64 + nt0 * 16 + (lane & 15);
      int col1 = h * 64 + nt1 * 16 + (lane & 15);
      size_t base = ((size_t)b * 49 + row) * 512;
      outw[base + col0] = (bf16_t)acc0[r];
      outw[base + col1] = (bf16_t)acc1[r];
    }
  }
}

// ---------------- 9. projection GEMM + bias -> [B, C, 7, 7] ----------------

__global__ __launch_bounds__(256) void mscn_proj_kernel(
    const bf16_t* __restrict__ outw, const bf16_t* __restrict__ WprojT,
    const float* __restrict__ bias, float* __restrict__ out) {
  int lane = threadIdx.x & 31, wave = threadIdx.x >> 5;
  int m0 = blockIdx.x * 128 + wave * 16;  // 25 row-blocks (padded rows zeroed)
  int n0 = blockIdx.y * 64;

  stage_tile(lds_dyn, WprojT + (size_t)n0 * 512, 64 * 512 * 2);

  v8f acc[4] = {zero8(), zero8(), zero8(), zero8()};
#pragma unroll 4
  for (int k0 = 0; k0 < 512; k0 += 32) {
    v16bf a = load_a_frag(outw, 512, m0, k0, lane);
#pragma unroll
    for (int nt = 0; nt < 4; ++nt) {
      v16bf b = load_b_frag(lds_dyn, 512, nt * 16, k0, lane);
      acc[nt] = wmma_bf16(a, b, acc[nt]);
    }
  }
#pragma unroll
  for (int nt = 0; nt < 4; ++nt) {
#pragma unroll
    for (int r = 0; r < 8; ++r) {
      int row = m0 + r + ((lane >> 4) << 3);
      if (row < RQ_) {
        int col = n0 + nt * 16 + (lane & 15);
        int b = row / 49, n = row % 49;
        out[((size_t)b * 512 + col) * 49 + n] = acc[nt][r] + bias[col];
      }
    }
  }
}

// ---------------------------------------------------------------------------

extern "C" void kernel_launch(void* const* d_in, const int* in_sizes, int n_in,
                              void* d_out, int out_size, void* d_ws, size_t ws_size,
                              hipStream_t stream) {
  const float* x      = (const float*)d_in[0];
  const float* y      = (const float*)d_in[1];
  const float* Wq     = (const float*)d_in[2];
  const float* Wkv    = (const float*)d_in[3];
  const float* Wproj  = (const float*)d_in[4];
  const float* bproj  = (const float*)d_in[5];
  const float* gamma  = (const float*)d_in[6];
  const float* beta   = (const float*)d_in[7];
  const float* attnw1 = (const float*)d_in[8];
  const float* attnw2 = (const float*)d_in[9];
  float* out = (float*)d_out;

  // workspace layout (pooled fp32 region is exactly reused for scores S)
  char* ws = (char*)d_ws;
  size_t off = 0;
  auto take = [&](size_t bytes) {
    void* p = ws + off;
    off = (off + bytes + 255) & ~(size_t)255;
    return p;
  };
  float*  pooled = (float*)take((size_t)RKV_ * C_ * 4);        // == S (same size)
  float*  S      = pooled;                                     // reuse after LN
  bf16_t* yn     = (bf16_t*)take((size_t)RKV_ * C_ * 2);
  bf16_t* Kbuf   = (bf16_t*)take((size_t)B_ * HEADS_ * N1_ * HD_ * 2);
  bf16_t* vT     = (bf16_t*)take((size_t)B_ * HEADS_ * HD_ * N1_ * 2);
  bf16_t* qws    = (bf16_t*)take((size_t)B_ * HEADS_ * 64 * 64 * 2);
  bf16_t* xs     = (bf16_t*)take((size_t)RQPAD_ * C_ * 2);
  bf16_t* outw   = (bf16_t*)take((size_t)RQPAD_ * C_ * 2);
  bf16_t* WqT    = (bf16_t*)take((size_t)512 * 512 * 2);
  bf16_t* WkvT   = (bf16_t*)take((size_t)1024 * 512 * 2);
  bf16_t* WprojT = (bf16_t*)take((size_t)512 * 512 * 2);
  float*  stats  = (float*)take((size_t)B_ * HEADS_ * 49 * 8 * 4);

  // zero padded regions (graph-capturable memsets)
  hipMemsetAsync(qws,  0, (size_t)B_ * HEADS_ * 64 * 64 * 2, stream);
  hipMemsetAsync(xs,   0, (size_t)RQPAD_ * C_ * 2, stream);
  hipMemsetAsync(outw, 0, (size_t)RQPAD_ * C_ * 2, stream);

  const int prep_total = 512 * 512 + 512 * 1024 + 512 * 512 + RQ_ * 512;
  mscn_prep_kernel<<<(prep_total + 255) / 256, 256, 0, stream>>>(
      Wq, Wkv, Wproj, x, WqT, WkvT, WprojT, xs);

  mscn_pool_kernel<<<B_ * C_, 256, 0, stream>>>(y, pooled);

  mscn_ln_kernel<<<RKV_, 256, 0, stream>>>(pooled, gamma, beta, yn);

  const size_t WTILE_LDS = 64 * 512 * 2;      // 64KB weight tile
  const size_t KTILE_LDS = (size_t)N1_ * 64 * 2;  // 98KB K / V^T tile

  mscn_kv_gemm_kernel<<<dim3(RKV_ / 128, 1024 / 64), 256, WTILE_LDS, stream>>>(
      yn, WkvT, Kbuf, vT);

  mscn_q_gemm_kernel<<<dim3(RQPAD_ / 128, 512 / 64), 256, WTILE_LDS, stream>>>(
      xs, WqT, qws);

  mscn_score_kernel<<<B_ * HEADS_, 256, KTILE_LDS, stream>>>(qws, Kbuf, S);

  mscn_stats_kernel<<<B_ * HEADS_ * 49, 256, 0, stream>>>(S, attnw1, attnw2, stats);

  mscn_av_kernel<<<B_ * HEADS_, 256, KTILE_LDS, stream>>>(S, stats, vT, outw);

  mscn_proj_kernel<<<dim3(RQPAD_ / 128, 512 / 64), 256, WTILE_LDS, stream>>>(
      outw, WprojT, bproj, out);

  (void)in_sizes; (void)n_in; (void)out_size; (void)ws_size;
}
